// CrossAttention_48344151884269
// MI455X (gfx1250) — matrix operand order repaired and checked by reading, not compiled, not run on hardware
//
#include <hip/hip_runtime.h>

#define NHEADS 16
#define DHEAD  64
#define DMODEL 1024
#define NQ     2048
#define NC     2048
#define BATCH  2

typedef __attribute__((ext_vector_type(16))) _Float16 v16h;
typedef __attribute__((ext_vector_type(8)))  _Float16 v8h;
typedef __attribute__((ext_vector_type(2)))  _Float16 h2;
typedef __attribute__((ext_vector_type(8)))  float    v8f;
typedef __attribute__((ext_vector_type(4)))  float    fx4;

__device__ __forceinline__ v8f wmma_f16(v16h a, v16h b, v8f c) {
  return __builtin_amdgcn_wmma_f32_16x16x32_f16(false, a, false, b, (short)0, c,
                                                false, false);
}

// Combine two 8-half runs into one 16-half WMMA operand register set.
__device__ __forceinline__ v16h combine16(v8h lo, v8h hi) {
  return __builtin_shufflevector(lo, hi, 0, 1, 2, 3, 4, 5, 6, 7,
                                 8, 9, 10, 11, 12, 13, 14, 15);
}

// B fragment (32x16 f16): lane holds col n=lane%16, element j -> k=16*half+j.
__device__ __forceinline__ v16h loadB(const _Float16* p) {
  return combine16(*(const v8h*)p, *(const v8h*)(p + 8));
}

// A fragment (16x32 f16). element j -> k = 16*half + j.
__device__ __forceinline__ v16h loadA_f16(const _Float16* p, int half) {
  return combine16(*(const v8h*)(p + 16 * half),
                   *(const v8h*)(p + 16 * half + 8));
}

// A fragment from f32 memory, converting to f16 on the fly.
__device__ __forceinline__ v16h loadA_f32(const float* p, int half) {
  fx4 a0 = *(const fx4*)(p + 16 * half);
  fx4 a1 = *(const fx4*)(p + 16 * half + 4);
  fx4 b0 = *(const fx4*)(p + 16 * half + 8);
  fx4 b1 = *(const fx4*)(p + 16 * half + 12);
  v16h r;
#pragma unroll
  for (int j = 0; j < 4; ++j) {
    r[j]      = (_Float16)a0[j];
    r[4 + j]  = (_Float16)a1[j];
    r[8 + j]  = (_Float16)b0[j];
    r[12 + j] = (_Float16)b1[j];
  }
  return r;
}

__device__ __forceinline__ float xor_sum16(float v) {
  v += __shfl_xor(v, 1, 32);
  v += __shfl_xor(v, 2, 32);
  v += __shfl_xor(v, 4, 32);
  v += __shfl_xor(v, 8, 32);
  return v;
}
__device__ __forceinline__ float xor_max16(float v) {
  v = fmaxf(v, __shfl_xor(v, 1, 32));
  v = fmaxf(v, __shfl_xor(v, 2, 32));
  v = fmaxf(v, __shfl_xor(v, 4, 32));
  v = fmaxf(v, __shfl_xor(v, 8, 32));
  return v;
}

// ---------------------------------------------------------------------------
// Fused projection GEMM: C[M=4096, N] = X[4096,1024] @ W[1024,N]
// mode 0: Q  -> per-head LayerNorm, *sqrt(64), f16 to Qn[b,h,nq,d]
// mode 1: KV -> cols <1024: K (LayerNorm) -> Kn ; cols >=1024: V -> Vt[d,nc]
// block = 128 threads (4 waves). Wave computes 16 rows x 64 cols (one head).
// Double-buffered LDS weight tile, one barrier per K-step.
// ---------------------------------------------------------------------------
__global__ __launch_bounds__(128) void proj_qkv_kernel(
    const float* __restrict__ X, const float* __restrict__ W,
    const float* __restrict__ ln_g, const float* __restrict__ ln_b,
    _Float16* __restrict__ Qn, _Float16* __restrict__ Kn,
    _Float16* __restrict__ Vt, int N, int mode) {
  __shared__ __attribute__((aligned(16))) _Float16 bt[2][64][40];  // [n][k]

  const int tid = threadIdx.x;
  const int lane = tid & 31, wave = tid >> 5;
  const int l16 = lane & 15, half = lane >> 4;
  const int n0 = blockIdx.x * 64;
  const int m0 = blockIdx.y * 64 + wave * 16;
  const float* arow = X + (size_t)(m0 + l16) * DMODEL;

  // Staging: thread owns k-pairs {2*kp0, 2*kp0+1} and {+16,+17}, n-quad q.
  const int q = tid & 15;
  const int kp0 = tid >> 4;  // 0..7
  fx4 r00, r01, r10, r11;    // hoisted staging registers (issue as a clause)

  auto loadW = [&](int kk) {
    const float* g0 = W + (size_t)(kk + 2 * kp0) * N + n0 + q * 4;
    const float* g1 = g0 + (size_t)16 * N;
    r00 = *(const fx4*)g0;
    r01 = *(const fx4*)(g0 + N);
    r10 = *(const fx4*)g1;
    r11 = *(const fx4*)(g1 + N);
  };
  auto storeW = [&](int buf) {
#pragma unroll
    for (int u = 0; u < 4; ++u) {
      h2 p0, p1;
      p0[0] = (_Float16)r00[u];
      p0[1] = (_Float16)r01[u];
      p1[0] = (_Float16)r10[u];
      p1[1] = (_Float16)r11[u];
      *(h2*)&bt[buf][q * 4 + u][2 * kp0] = p0;       // k, k+1
      *(h2*)&bt[buf][q * 4 + u][2 * kp0 + 16] = p1;  // k+16, k+17
    }
  };

  v8f c0 = {}, c1 = {}, c2 = {}, c3 = {};
  loadW(0);
  storeW(0);
  __syncthreads();

  for (int t = 0; t < DMODEL / 32; ++t) {
    const int kk = t * 32;
    const int buf = t & 1;
    if (kk + 32 < DMODEL) loadW(kk + 32);  // overlap next tile with WMMAs
    if (kk + 64 < DMODEL)                  // deeper prefetch (global_prefetch)
      __builtin_prefetch(W + (size_t)(kk + 64 + (tid >> 2)) * N + n0 +
                             (tid & 3) * 16, 0, 1);
    v16h a = loadA_f32(arow + kk, half);
    c0 = wmma_f16(a, loadB(&bt[buf][0 + l16][16 * half]), c0);
    c1 = wmma_f16(a, loadB(&bt[buf][16 + l16][16 * half]), c1);
    c2 = wmma_f16(a, loadB(&bt[buf][32 + l16][16 * half]), c2);
    c3 = wmma_f16(a, loadB(&bt[buf][48 + l16][16 * half]), c3);
    if (kk + 32 < DMODEL) storeW(buf ^ 1);  // other buffer: no conflict
    __syncthreads();
  }

  v8f cc[4] = {c0, c1, c2, c3};
  const bool isV = (mode == 1) && (n0 >= DMODEL);
  const int h = isV ? (n0 - DMODEL) / DHEAD : (n0 / DHEAD) % NHEADS;
  const float qscale = (mode == 0) ? 8.0f : 1.0f;  // sqrt(head_dim)

  if (!isV) {
    float gv[4], bv[4];
#pragma unroll
    for (int ch = 0; ch < 4; ++ch) {
      gv[ch] = ln_g[ch * 16 + l16];
      bv[ch] = ln_b[ch * 16 + l16];
    }
#pragma unroll
    for (int i = 0; i < 8; ++i) {
      int mg = m0 + i + 8 * half;  // global row (b * 2048 + token)
      int b = mg >> 11;
      int tk = mg & 2047;
      // LayerNorm over this wave's 64 columns (= one head).
      float s = cc[0][i] + cc[1][i] + cc[2][i] + cc[3][i];
      float mu = xor_sum16(s) * (1.0f / 64.0f);
      float d0 = cc[0][i] - mu, d1 = cc[1][i] - mu;
      float d2 = cc[2][i] - mu, d3 = cc[3][i] - mu;
      float vs = xor_sum16(d0 * d0 + d1 * d1 + d2 * d2 + d3 * d3);
      float rstd = rsqrtf(vs * (1.0f / 64.0f) + 1e-5f);
      _Float16* dst = (mode == 0 ? Qn : Kn) +
                      ((size_t)((b * NHEADS + h) * NQ + tk)) * DHEAD;
#pragma unroll
      for (int ch = 0; ch < 4; ++ch) {
        float y = ((cc[ch][i] - mu) * rstd * gv[ch] + bv[ch]) * qscale;
        dst[ch * 16 + l16] = (_Float16)y;
      }
    }
  } else {
    // V: d-major store; each lane's 8 rows are consecutive tokens at fixed d
    // -> one 16B store per chunk (global_store_b128).
    int mg0 = m0 + 8 * half;
    int b = mg0 >> 11;
    int t0 = mg0 & 2047;
#pragma unroll
    for (int ch = 0; ch < 4; ++ch) {
      int d = ch * 16 + l16;
      v8h pk;
#pragma unroll
      for (int i = 0; i < 8; ++i) pk[i] = (_Float16)cc[ch][i];
      *(v8h*)(Vt + ((size_t)((b * NHEADS + h) * DHEAD + d)) * NC + t0) = pk;
    }
  }
}

// ---------------------------------------------------------------------------
// Flash attention: grid (Nq/64, B*H), 4 waves. Double-buffered K/V LDS tiles
// shared by all waves; wave owns 16 query rows; online softmax in f32.
// ---------------------------------------------------------------------------
__global__ __launch_bounds__(128) void attn_kernel(
    const _Float16* __restrict__ Qn, const _Float16* __restrict__ Kn,
    const _Float16* __restrict__ Vt, _Float16* __restrict__ Of) {
  __shared__ __attribute__((aligned(16))) _Float16 klds[2][32][72];  // [key][d]
  __shared__ __attribute__((aligned(16))) _Float16 vlds[2][64][40];  // [d][key]
  __shared__ __attribute__((aligned(16))) _Float16 pbuf[4][16][40];

  const int tid = threadIdx.x;
  const int lane = tid & 31, wave = tid >> 5;
  const int l16 = lane & 15, half = lane >> 4;
  const int bh = blockIdx.y;
  const int b = bh >> 4, h = bh & 15;
  const int m0 = blockIdx.x * 64 + wave * 16;

  const _Float16* qrow = Qn + ((size_t)bh * NQ + m0 + l16) * DHEAD;
  v16h aq0 = loadA_f16(qrow, half);       // d = 0..31
  v16h aq1 = loadA_f16(qrow + 32, half);  // d = 32..63

  v8f o0 = {}, o1 = {}, o2 = {}, o3 = {};
  float mi[8], li[8];
#pragma unroll
  for (int i = 0; i < 8; ++i) {
    mi[i] = -3.0e38f;
    li[i] = 0.0f;
  }

  const _Float16* kbase = Kn + (size_t)bh * NC * DHEAD;
  const _Float16* vbase = Vt + (size_t)bh * DHEAD * NC;

  // Staging assignments (256 octet-copies per tile, 2 per thread per array).
  const int lin0 = tid, lin1 = tid + 128;
  v8h rk0, rk1, rv0, rv1;  // hoisted: all 4 global loads issue together
  auto loadKV = [&](int kb) {
    rk0 = *(const v8h*)(kbase + (size_t)(kb + (lin0 >> 3)) * DHEAD +
                        (lin0 & 7) * 8);
    rk1 = *(const v8h*)(kbase + (size_t)(kb + (lin1 >> 3)) * DHEAD +
                        (lin1 & 7) * 8);
    rv0 = *(const v8h*)(vbase + (size_t)(lin0 >> 2) * NC + kb + (lin0 & 3) * 8);
    rv1 = *(const v8h*)(vbase + (size_t)(lin1 >> 2) * NC + kb + (lin1 & 3) * 8);
  };
  auto storeKV = [&](int buf) {
    *(v8h*)&klds[buf][lin0 >> 3][(lin0 & 7) * 8] = rk0;
    *(v8h*)&klds[buf][lin1 >> 3][(lin1 & 7) * 8] = rk1;
    *(v8h*)&vlds[buf][lin0 >> 2][(lin0 & 3) * 8] = rv0;
    *(v8h*)&vlds[buf][lin1 >> 2][(lin1 & 3) * 8] = rv1;
  };

  loadKV(0);
  storeKV(0);
  __syncthreads();

  for (int kb = 0; kb < NC; kb += 32) {
    const int buf = (kb >> 5) & 1;
    if (kb + 32 < NC) loadKV(kb + 32);  // overlap with this tile's math
    if (kb + 64 < NC)
      __builtin_prefetch(kbase + (size_t)(kb + 64) * DHEAD + tid * 16, 0, 1);

    // S = (Q*8,LN) @ K^T for 32 keys, as two 16x16 f32 tiles (from LDS).
    v8f s0 = {}, s1 = {};
    s0 = wmma_f16(aq0, loadB(&klds[buf][0 + l16][16 * half]), s0);
    s0 = wmma_f16(aq1, loadB(&klds[buf][0 + l16][32 + 16 * half]), s0);
    s1 = wmma_f16(aq0, loadB(&klds[buf][16 + l16][16 * half]), s1);
    s1 = wmma_f16(aq1, loadB(&klds[buf][16 + l16][32 + 16 * half]), s1);

    // Online softmax update per query row; write P (f16) to per-wave LDS.
#pragma unroll
    for (int i = 0; i < 8; ++i) {
      float mx = xor_max16(fmaxf(s0[i], s1[i]));
      float mnew = fmaxf(mi[i], mx);
      float alpha = __expf(mi[i] - mnew);
      float p0 = __expf(s0[i] - mnew);
      float p1 = __expf(s1[i] - mnew);
      float rs = xor_sum16(p0 + p1);
      li[i] = li[i] * alpha + rs;
      mi[i] = mnew;
      o0[i] *= alpha;
      o1[i] *= alpha;
      o2[i] *= alpha;
      o3[i] *= alpha;
      pbuf[wave][i + 8 * half][l16] = (_Float16)p0;
      pbuf[wave][i + 8 * half][16 + l16] = (_Float16)p1;
    }

    // C-layout -> A-layout transpose through per-wave LDS (in-order DS).
    v16h ap = loadA_f16(&pbuf[wave][l16][0], half);
    o0 = wmma_f16(ap, loadB(&vlds[buf][0 + l16][16 * half]), o0);
    o1 = wmma_f16(ap, loadB(&vlds[buf][16 + l16][16 * half]), o1);
    o2 = wmma_f16(ap, loadB(&vlds[buf][32 + l16][16 * half]), o2);
    o3 = wmma_f16(ap, loadB(&vlds[buf][48 + l16][16 * half]), o3);

    if (kb + 32 < NC) storeKV(buf ^ 1);  // other buffer: no conflict
    __syncthreads();
  }

  // Normalize and store f16 output at [b, q, h*64 + d].
#pragma unroll
  for (int i = 0; i < 8; ++i) {
    float inv = 1.0f / li[i];
    int qq = m0 + i + 8 * half;
    _Float16* dst = Of + ((size_t)(b * NQ + qq)) * DMODEL + h * DHEAD;
    dst[0 + l16] = (_Float16)(o0[i] * inv);
    dst[16 + l16] = (_Float16)(o1[i] * inv);
    dst[32 + l16] = (_Float16)(o2[i] * inv);
    dst[48 + l16] = (_Float16)(o3[i] * inv);
  }
}

// ---------------------------------------------------------------------------
// Output projection: out[4096,1024] = Of(f16) @ Wp + bp, f32 result.
// Same double-buffered structure as proj_qkv.
// ---------------------------------------------------------------------------
__global__ __launch_bounds__(128) void proj_out_kernel(
    const _Float16* __restrict__ Of, const float* __restrict__ Wp,
    const float* __restrict__ bp, float* __restrict__ out) {
  __shared__ __attribute__((aligned(16))) _Float16 bt[2][64][40];

  const int tid = threadIdx.x;
  const int lane = tid & 31, wave = tid >> 5;
  const int l16 = lane & 15, half = lane >> 4;
  const int n0 = blockIdx.x * 64;
  const int m0 = blockIdx.y * 64 + wave * 16;
  const _Float16* arow = Of + (size_t)(m0 + l16) * DMODEL;

  const int q = tid & 15;
  const int kp0 = tid >> 4;
  fx4 r00, r01, r10, r11;

  auto loadW = [&](int kk) {
    const float* g0 = Wp + (size_t)(kk + 2 * kp0) * DMODEL + n0 + q * 4;
    const float* g1 = g0 + (size_t)16 * DMODEL;
    r00 = *(const fx4*)g0;
    r01 = *(const fx4*)(g0 + DMODEL);
    r10 = *(const fx4*)g1;
    r11 = *(const fx4*)(g1 + DMODEL);
  };
  auto storeW = [&](int buf) {
#pragma unroll
    for (int u = 0; u < 4; ++u) {
      h2 p0, p1;
      p0[0] = (_Float16)r00[u];
      p0[1] = (_Float16)r01[u];
      p1[0] = (_Float16)r10[u];
      p1[1] = (_Float16)r11[u];
      *(h2*)&bt[buf][q * 4 + u][2 * kp0] = p0;
      *(h2*)&bt[buf][q * 4 + u][2 * kp0 + 16] = p1;
    }
  };

  v8f c0 = {}, c1 = {}, c2 = {}, c3 = {};
  loadW(0);
  storeW(0);
  __syncthreads();

  for (int t = 0; t < DMODEL / 32; ++t) {
    const int kk = t * 32;
    const int buf = t & 1;
    if (kk + 32 < DMODEL) loadW(kk + 32);
    if (kk + 64 < DMODEL)
      __builtin_prefetch(Wp + (size_t)(kk + 64 + (tid >> 2)) * DMODEL + n0 +
                             (tid & 3) * 16, 0, 1);
    v16h a = loadA_f16(arow + kk, half);
    c0 = wmma_f16(a, loadB(&bt[buf][0 + l16][16 * half]), c0);
    c1 = wmma_f16(a, loadB(&bt[buf][16 + l16][16 * half]), c1);
    c2 = wmma_f16(a, loadB(&bt[buf][32 + l16][16 * half]), c2);
    c3 = wmma_f16(a, loadB(&bt[buf][48 + l16][16 * half]), c3);
    if (kk + 32 < DMODEL) storeW(buf ^ 1);
    __syncthreads();
  }

  v8f cc[4] = {c0, c1, c2, c3};
  float bpv[4];
#pragma unroll
  for (int ch = 0; ch < 4; ++ch) bpv[ch] = bp[n0 + ch * 16 + l16];

#pragma unroll
  for (int i = 0; i < 8; ++i) {
    int mg = m0 + i + 8 * half;
    float* dst = out + (size_t)mg * DMODEL + n0;
#pragma unroll
    for (int ch = 0; ch < 4; ++ch)
      dst[ch * 16 + l16] = cc[ch][i] + bpv[ch];
  }
}

// ---------------------------------------------------------------------------
extern "C" void kernel_launch(void* const* d_in, const int* in_sizes, int n_in,
                              void* d_out, int out_size, void* d_ws,
                              size_t ws_size, hipStream_t stream) {
  const float* xq = (const float*)d_in[0];
  const float* xc = (const float*)d_in[1];
  const float* Wq = (const float*)d_in[2];
  const float* Wkv = (const float*)d_in[3];
  const float* Wp = (const float*)d_in[4];
  const float* bp = (const float*)d_in[5];
  const float* ln_g = (const float*)d_in[6];
  const float* ln_b = (const float*)d_in[7];
  float* out = (float*)d_out;

  char* ws = (char*)d_ws;
  size_t off = 0;
  auto carve = [&](size_t bytes) {
    void* p = ws + off;
    off = (off + bytes + 255) & ~(size_t)255;
    return p;
  };
  const size_t elems = (size_t)BATCH * NHEADS * NQ * DHEAD;  // 4M halves
  _Float16* Qn = (_Float16*)carve(elems * sizeof(_Float16));
  _Float16* Kn = (_Float16*)carve(elems * sizeof(_Float16));
  _Float16* Vt = (_Float16*)carve(elems * sizeof(_Float16));
  _Float16* Of = (_Float16*)carve((size_t)BATCH * NQ * DMODEL * sizeof(_Float16));

  dim3 blk(128);
  // Q projection + per-head LN + *sqrt(d): M=4096, N=1024
  proj_qkv_kernel<<<dim3(DMODEL / 64, (BATCH * NQ) / 64), blk, 0, stream>>>(
      xq, Wq, ln_g, ln_b, Qn, nullptr, nullptr, DMODEL, 0);
  // KV projection (+LN for K, transpose-store V): M=4096, N=2048
  proj_qkv_kernel<<<dim3((2 * DMODEL) / 64, (BATCH * NC) / 64), blk, 0,
                    stream>>>(xc, Wkv, ln_g, ln_b, nullptr, Kn, Vt,
                              2 * DMODEL, 1);
  // Flash attention: grid (Nq/64, B*H)
  attn_kernel<<<dim3(NQ / 64, BATCH * NHEADS), blk, 0, stream>>>(Qn, Kn, Vt,
                                                                 Of);
  // Output projection + bias
  proj_out_kernel<<<dim3(DMODEL / 64, (BATCH * NQ) / 64), blk, 0, stream>>>(
      Of, Wp, bp, out);
}
